// ContrastivePairEncoder_50843822850391
// MI455X (gfx1250) — compile-verified, matrix operand order checked by reading, not run on hardware
//
#include <hip/hip_runtime.h>
#include <hip/hip_bf16.h>
#include <math.h>

// ---------------- problem constants ----------------
#define B_  128
#define E_  256
#define O_  256
#define T_  32
#define H_  768
#define P_  256
#define TEMP_ 0.07f

typedef __bf16 bf16;
typedef __attribute__((ext_vector_type(16))) __bf16 v16bf;
typedef __attribute__((ext_vector_type(8)))  __bf16 v8bf;
typedef __attribute__((ext_vector_type(4)))  __bf16 v4bf;
typedef __attribute__((ext_vector_type(8)))  float  v8f;
typedef __attribute__((ext_vector_type(4)))  float  f32x4;

#define MT 16   // rows per block (M tile)
#define KT 32   // K depth per WMMA

// ---------------- A fragment from row-major bf16 LDS tile ----------------
// ISA 7.12.2: lanes 0-15 row M=lane, K={kc+0..7, kc+16..23};
//             lanes 16-31 row M=lane-16, K={kc+8..15, kc+24..31}
// Both 8-element runs are contiguous & 16B-aligned -> 2x ds_load_b128.
__device__ __forceinline__ v16bf load_a_frag(const bf16* __restrict__ src, int ldw,
                                             int kc, int lane) {
  int m   = lane & 15;
  int off = (lane >> 4) << 3;
  const bf16* base = src + m * ldw + kc + off;
  v8bf lo = *(const v8bf*)(base);
  v8bf hi = *(const v8bf*)(base + 16);
  return __builtin_shufflevector(lo, hi, 0, 1, 2, 3, 4, 5, 6, 7,
                                         8, 9, 10, 11, 12, 13, 14, 15);
}

// ---------------- ping-pong pipelined K-loop over one wave's n-tiles --------
// A from LDS (row-major), B from fragment-major global weights.
// Two stable buffer sets (a0/b0, a1/b1) alternate roles: no rotating register
// copies, prefetch distance of one full WMMA group.
template <int TPW>
__device__ __forceinline__ void gemm_tiles(const bf16* __restrict__ sA,
                                           const v16bf* __restrict__ Wv,
                                           int ntilesTotal, int wave, int lane,
                                           v8f* acc) {
  constexpr int NK = H_ / KT;   // 24 (even)
  const int tbase = wave * TPW;

  v16bf a0, a1, b0[TPW], b1[TPW];

  a0 = load_a_frag(sA, H_, 0, lane);
#pragma unroll
  for (int j = 0; j < TPW; ++j)
    b0[j] = Wv[(size_t)(tbase + j) * 32 + lane];
  a1 = load_a_frag(sA, H_, KT, lane);
#pragma unroll
  for (int j = 0; j < TPW; ++j)
    b1[j] = Wv[((size_t)ntilesTotal + tbase + j) * 32 + lane];

  for (int kcI = 0; kcI < NK - 2; kcI += 2) {
    // compute chunk kcI, refill b0 with chunk kcI+2
#pragma unroll
    for (int j = 0; j < TPW; ++j)
      acc[j] = __builtin_amdgcn_wmma_f32_16x16x32_bf16(
          false, a0, false, b0[j], (short)0, acc[j], false, false);
    a0 = load_a_frag(sA, H_, (kcI + 2) * KT, lane);
#pragma unroll
    for (int j = 0; j < TPW; ++j)
      b0[j] = Wv[((size_t)(kcI + 2) * ntilesTotal + tbase + j) * 32 + lane];

    // compute chunk kcI+1, refill b1 with chunk kcI+3
#pragma unroll
    for (int j = 0; j < TPW; ++j)
      acc[j] = __builtin_amdgcn_wmma_f32_16x16x32_bf16(
          false, a1, false, b1[j], (short)0, acc[j], false, false);
    a1 = load_a_frag(sA, H_, (kcI + 3) * KT, lane);
#pragma unroll
    for (int j = 0; j < TPW; ++j)
      b1[j] = Wv[((size_t)(kcI + 3) * ntilesTotal + tbase + j) * 32 + lane];
  }
  // epilogue: chunks NK-2 (a0/b0) and NK-1 (a1/b1)
#pragma unroll
  for (int j = 0; j < TPW; ++j)
    acc[j] = __builtin_amdgcn_wmma_f32_16x16x32_bf16(
        false, a0, false, b0[j], (short)0, acc[j], false, false);
#pragma unroll
  for (int j = 0; j < TPW; ++j)
    acc[j] = __builtin_amdgcn_wmma_f32_16x16x32_bf16(
        false, a1, false, b1[j], (short)0, acc[j], false, false);
}

// ---------------- weight convert f32 -> bf16, fragment-major ----------------
// dst layout: [kchunk][ntile][lane][16 bf16]: a wave's B-fragment for one
// 16x16 n-tile is 32 consecutive 32-byte packets (b128-pair per lane).
__global__ void cvt_frag_kernel(const float* __restrict__ src, bf16* __restrict__ dst,
                                int K, int N) {
  int i = blockIdx.x * 256 + threadIdx.x;
  if (i >= K * N) return;
  int k = i / N, c = i - k * N;
  int kchunk = k >> 5, rl = k & 31;
  int tile = c >> 4, m = c & 15;
  int lane = m + ((rl >> 3) & 1) * 16;
  int idx  = ((rl >> 4) << 3) | (rl & 7);
  int ntiles = N >> 4;
  dst[(((size_t)(kchunk * ntiles + tile) * 32) + lane) * 16 + idx] = (bf16)src[i];
}

// ---------------- fused 2-layer MLP + L2 normalize ----------------
// Block: 256 threads (8 wave32s), produces 16 normalized rows of [P_].
// pairs==nullptr : rows r0..r0+15 of X[b] (opinions)
// pairs!=nullptr : gathered rows entity_reprs[b, pairs[(b*outRows+r)*2]]
__global__ __launch_bounds__(256) void encode_kernel(
    const float* __restrict__ X, const int* __restrict__ pairs,
    int inRowsPerBatch, int outRowsPerBatch,
    const bf16* __restrict__ W1f, const float* __restrict__ b1,
    const bf16* __restrict__ W2f, const float* __restrict__ b2,
    float* __restrict__ out) {
  extern __shared__ char smem[];
  bf16*  sX = (bf16*)smem;            // MT x H_   (24 KB)
  bf16*  sH = sX + MT * H_;           // MT x H_   (24 KB)
  float* sZ = (float*)(sH + MT * H_); // MT x P_   (16 KB)
  float* sR = sZ + MT * P_;           // 256 floats

  const int tid  = threadIdx.x;
  const int lane = tid & 31;
  const int wave = tid >> 5;

  const int tilesPerBatch = outRowsPerBatch / MT;
  const int b  = blockIdx.x / tilesPerBatch;
  const int r0 = (blockIdx.x % tilesPerBatch) * MT;

  // ---- stage input rows (optional gather): float4 load -> 4x bf16 store ----
  for (int i = tid; i < MT * (H_ / 4); i += 256) {
    int r  = i / (H_ / 4);
    int c4 = i - r * (H_ / 4);
    int srcRow = pairs ? pairs[((size_t)b * outRowsPerBatch + r0 + r) * 2] : (r0 + r);
    f32x4 v = *(const f32x4*)(X + ((size_t)b * inRowsPerBatch + srcRow) * H_ + c4 * 4);
    v4bf o; o[0] = (bf16)v[0]; o[1] = (bf16)v[1]; o[2] = (bf16)v[2]; o[3] = (bf16)v[3];
    *(v4bf*)(sX + r * H_ + c4 * 4) = o;
  }
  __syncthreads();

  const int mrow = (lane >> 4) * 8;  // C-frag: VGPR r -> M = r + mrow
  const int ncol = lane & 15;        //         N = ntile + ncol

  const v16bf* W1v = (const v16bf*)W1f;  // [kchunk][48][32]
  const v16bf* W2v = (const v16bf*)W2f;  // [kchunk][16][32]

  // ================= layer 1: h = relu(X @ W1 + b1), 48 n-tiles =================
  v8f acc1[6];
#pragma unroll
  for (int j = 0; j < 6; ++j) acc1[j] = (v8f){0, 0, 0, 0, 0, 0, 0, 0};
  gemm_tiles<6>(sX, W1v, 48, wave, lane, acc1);

  // bias + relu -> bf16 h in LDS (each wave owns a disjoint n-range)
#pragma unroll
  for (int j = 0; j < 6; ++j) {
    int n0 = (wave * 6 + j) * 16;
    float bias = b1[n0 + ncol];
#pragma unroll
    for (int r = 0; r < 8; ++r) {
      float v = acc1[j][r] + bias;
      sH[(r + mrow) * H_ + n0 + ncol] = (bf16)(v > 0.f ? v : 0.f);
    }
  }
  __syncthreads();

  // ================= layer 2: z = h @ W2 + b2, 16 n-tiles =================
  v8f acc2[2];
#pragma unroll
  for (int j = 0; j < 2; ++j) acc2[j] = (v8f){0, 0, 0, 0, 0, 0, 0, 0};
  gemm_tiles<2>(sH, W2v, 16, wave, lane, acc2);

  // bias -> f32 z in LDS
#pragma unroll
  for (int j = 0; j < 2; ++j) {
    int n0 = (wave * 2 + j) * 16;
    float bias = b2[n0 + ncol];
#pragma unroll
    for (int r = 0; r < 8; ++r)
      sZ[(r + mrow) * P_ + n0 + ncol] = acc2[j][r] + bias;
  }
  __syncthreads();

  // ---- row L2 norms: 16 threads per row, 16-wide segments ----
  {
    int row = tid >> 4, seg = tid & 15;
    const f32x4* zr = (const f32x4*)(sZ + row * P_ + seg * 16);
    float s = 0.f;
#pragma unroll
    for (int i = 0; i < 4; ++i) {
      f32x4 v = zr[i];
      s += v[0] * v[0] + v[1] * v[1] + v[2] * v[2] + v[3] * v[3];
    }
    sR[row * 16 + seg] = s;
  }
  __syncthreads();
  if (tid < MT) {
    float s = 0.f;
#pragma unroll
    for (int i = 0; i < 16; ++i) s += sR[tid * 16 + i];
    sR[tid * 16] = 1.f / fmaxf(sqrtf(s), 1e-12f);
  }
  __syncthreads();
  {
    int row = tid >> 4, seg = tid & 15;
    float inv = sR[row * 16];
    const f32x4* zr = (const f32x4*)(sZ + row * P_ + seg * 16);
    f32x4* orow = (f32x4*)(out + ((size_t)b * outRowsPerBatch + r0 + row) * P_ + seg * 16);
#pragma unroll
    for (int i = 0; i < 4; ++i) {
      f32x4 v = zr[i];
      orow[i] = (f32x4){v[0] * inv, v[1] * inv, v[2] * inv, v[3] * inv};
    }
  }
}

// ---------------- per-pair InfoNCE row term ----------------
__global__ __launch_bounds__(256) void loss_rows_kernel(
    const float* __restrict__ epn,   // [B,T,P] normalized gathered entities
    const float* __restrict__ opn,   // [B,O,P] normalized opinions
    const int* __restrict__ pairs,   // [B,T,2]
    float* __restrict__ loss_buf) {  // [B*T]
  __shared__ float red[256];
  __shared__ float s_pos;

  const int bt = blockIdx.x;
  const int bb = bt / T_;
  const int o  = threadIdx.x;

  const f32x4* e = (const f32x4*)(epn + (size_t)bt * P_);
  const f32x4* v = (const f32x4*)(opn + ((size_t)bb * O_ + o) * P_);
  float dot = 0.f;
#pragma unroll 8
  for (int i = 0; i < P_ / 4; ++i) {
    f32x4 a = e[i], c = v[i];
    dot += a[0] * c[0] + a[1] * c[1] + a[2] * c[2] + a[3] * c[3];
  }
  float s = dot * (1.0f / TEMP_);

  red[o] = s;
  __syncthreads();
  for (int st = 128; st > 0; st >>= 1) {
    if (o < st) red[o] = fmaxf(red[o], red[o + st]);
    __syncthreads();
  }
  float smax = red[0];
  __syncthreads();
  red[o] = expf(s - smax);
  __syncthreads();
  for (int st = 128; st > 0; st >>= 1) {
    if (o < st) red[o] += red[o + st];
    __syncthreads();
  }
  int oidx = pairs[(size_t)bt * 2 + 1];
  if (o == oidx) s_pos = s;
  __syncthreads();
  if (o == 0) loss_buf[bt] = (smax + logf(red[0])) - s_pos;
}

// ---------------- deterministic final mean ----------------
__global__ __launch_bounds__(256) void reduce_mean_kernel(
    const float* __restrict__ loss_buf, float* __restrict__ out, int n) {
  __shared__ float red[256];
  float s = 0.f;
  for (int i = threadIdx.x; i < n; i += 256) s += loss_buf[i];
  red[threadIdx.x] = s;
  __syncthreads();
  for (int st = 128; st > 0; st >>= 1) {
    if (threadIdx.x < st) red[threadIdx.x] += red[threadIdx.x + st];
    __syncthreads();
  }
  if (threadIdx.x == 0) out[0] = red[0] / (float)n;
}

// ---------------- host launcher ----------------
extern "C" void kernel_launch(void* const* d_in, const int* in_sizes, int n_in,
                              void* d_out, int out_size, void* d_ws, size_t ws_size,
                              hipStream_t stream) {
  (void)in_sizes; (void)n_in; (void)out_size; (void)ws_size;

  const float* entity  = (const float*)d_in[0];
  const float* opinion = (const float*)d_in[1];
  const int*   pairs   = (const int*)d_in[2];
  const float* eW1 = (const float*)d_in[3];
  const float* eb1 = (const float*)d_in[4];
  const float* eW2 = (const float*)d_in[5];
  const float* eb2 = (const float*)d_in[6];
  const float* oW1 = (const float*)d_in[7];
  const float* ob1 = (const float*)d_in[8];
  const float* oW2 = (const float*)d_in[9];
  const float* ob2 = (const float*)d_in[10];

  // -------- workspace carve (all 32B-aligned sizes) --------
  char* ws = (char*)d_ws;
  bf16* eW1f = (bf16*)ws;              ws += (size_t)H_ * H_ * sizeof(bf16);
  bf16* eW2f = (bf16*)ws;              ws += (size_t)H_ * P_ * sizeof(bf16);
  bf16* oW1f = (bf16*)ws;              ws += (size_t)H_ * H_ * sizeof(bf16);
  bf16* oW2f = (bf16*)ws;              ws += (size_t)H_ * P_ * sizeof(bf16);
  float* opn = (float*)ws;             ws += (size_t)B_ * O_ * P_ * sizeof(float);
  float* epn = (float*)ws;             ws += (size_t)B_ * T_ * P_ * sizeof(float);
  float* loss_buf = (float*)ws;        ws += (size_t)B_ * T_ * sizeof(float);

  // -------- 1) weight conversion to fragment-major bf16 --------
  {
    int n1 = H_ * H_, n2 = H_ * P_;
    cvt_frag_kernel<<<(n1 + 255) / 256, 256, 0, stream>>>(eW1, eW1f, H_, H_);
    cvt_frag_kernel<<<(n2 + 255) / 256, 256, 0, stream>>>(eW2, eW2f, H_, P_);
    cvt_frag_kernel<<<(n1 + 255) / 256, 256, 0, stream>>>(oW1, oW1f, H_, H_);
    cvt_frag_kernel<<<(n2 + 255) / 256, 256, 0, stream>>>(oW2, oW2f, H_, P_);
  }

  // -------- 2) encoders --------
  const size_t smemBytes =
      (size_t)(2 * MT * H_) * sizeof(bf16) +          // sX + sH
      (size_t)(MT * P_ + 256) * sizeof(float);        // sZ + sR  (~65.5 KB)

  // opinions: all B*O rows
  encode_kernel<<<B_ * (O_ / MT), 256, smemBytes, stream>>>(
      opinion, nullptr, O_, O_, oW1f, ob1, oW2f, ob2, opn);
  // entities: only the gathered B*T rows
  encode_kernel<<<B_ * (T_ / MT), 256, smemBytes, stream>>>(
      entity, pairs, E_, T_, eW1f, eb1, eW2f, eb2, epn);

  // -------- 3) loss rows + deterministic mean --------
  loss_rows_kernel<<<B_ * T_, 256, 0, stream>>>(epn, opn, pairs, loss_buf);
  reduce_mean_kernel<<<1, 256, 0, stream>>>(loss_buf, (float*)d_out, B_ * T_);
}
